// VariationalLSTMModule_35476429865300
// MI455X (gfx1250) — matrix operand order repaired and checked by reading, not compiled, not run on hardware
//
#include <hip/hip_runtime.h>
#include <stdint.h>

// Problem constants (match reference)
constexpr int L = 2, B = 64, T = 128, H = 1024, O = 10000;
constexpr int BH = B * H;

// LDS B-tile geometry: 64 columns x 64 K-elems (bf16), padded column stride 144 B
constexpr int KC = 64;                  // K elements per chunk
constexpr int COL_STRIDE = 144;         // bytes per column (128 data + 16 pad), 16B aligned
constexpr int BTILE_BYTES = 64 * COL_STRIDE;   // 9216 B per buffer

typedef __attribute__((ext_vector_type(16))) __bf16 v16bf;
typedef __attribute__((ext_vector_type(8)))  float  v8f;

struct U32x8 { uint4 lo; uint4 hi; };

__device__ __forceinline__ uint16_t f2bf(float f) {
    uint32_t u = __builtin_bit_cast(uint32_t, f);
    uint32_t r = (u + 0x7FFFu + ((u >> 16) & 1u)) >> 16;   // round-to-nearest-even
    return (uint16_t)r;
}

__device__ __forceinline__ float sigmoidf(float x) { return 1.0f / (1.0f + __expf(-x)); }

__device__ __forceinline__ void wait_async0() {
#if defined(__has_builtin) && __has_builtin(__builtin_amdgcn_s_wait_asynccnt)
    __builtin_amdgcn_s_wait_asynccnt(0);
#else
    asm volatile("s_wait_asynccnt 0" ::: "memory");
#endif
}

// ---- WMMA fragment loaders (wave32, 16-bit element layouts per CDNA5 ISA 7.12.2) ----
// A: 16x32 (MxK), row-major source, lda elements.
//   lanes 0-15 : row=lane,    K = k0+0..7  and k0+16..23
//   lanes 16-31: row=lane-16, K = k0+8..15 and k0+24..31
__device__ __forceinline__ v16bf load_frag_a(const uint16_t* __restrict__ base, int lda,
                                             int k0, int lane) {
    int row = lane & 15, sel = lane >> 4;
    const uint16_t* p = base + (size_t)row * lda + k0 + sel * 8;
    U32x8 r;
    r.lo = *reinterpret_cast<const uint4*>(p);
    r.hi = *reinterpret_cast<const uint4*>(p + 16);
    return __builtin_bit_cast(v16bf, r);
}

// B: 32x16 (KxN) fragment from LDS tile (column c at buf + c*COL_STRIDE, K contiguous).
//   lanes 0-15 : col=colbase+lane,    K = ks+0..7  and ks+16..23
//   lanes 16-31: col=colbase+lane-16, K = ks+8..15 and ks+24..31
__device__ __forceinline__ v16bf lds_frag_b(const uint8_t* __restrict__ buf,
                                            int colbase, int ks, int lane) {
    int col = colbase + (lane & 15), sel = lane >> 4;
    const uint8_t* p = buf + col * COL_STRIDE + (ks + sel * 8) * 2;
    U32x8 r;
    r.lo = *reinterpret_cast<const uint4*>(p);
    r.hi = *reinterpret_cast<const uint4*>(p + 32);
    return __builtin_bit_cast(v16bf, r);
}

// Cooperative async copy of one B tile: rows (N0..N0+63) of WT[maxrow+1][ldb],
// K slice [k0, k0+64) -> LDS buffer. 256 threads x 2 x 16B = 8 KB payload.
__device__ __forceinline__ void async_copy_b_tile(const uint16_t* __restrict__ src,
                                                  int N0, int k0, int maxrow, int ldb,
                                                  uint8_t* ldsbuf, int tid) {
    #pragma unroll
    for (int s = 0; s < 2; ++s) {
        int u = tid + s * 256;
        int c = u >> 3, o8 = u & 7;
        int row = N0 + c; if (row > maxrow) row = maxrow;
        uint64_t g = (uint64_t)(uintptr_t)(src + (size_t)row * ldb + k0 + o8 * 8);
        uint32_t l = (uint32_t)(uintptr_t)(ldsbuf + c * COL_STRIDE + o8 * 16);
        asm volatile("global_load_async_to_lds_b128 %0, %1, off"
                     :: "v"(l), "v"(g) : "memory");
    }
}

// One 16-chunk K-sweep (one A/B matrix pair) with double-buffered async B staging.
// A is a phase-constant pointer (kernel-arg derived) so loads stay GLOBAL, not FLAT.
__device__ __forceinline__ void gemm_phase(
    const uint16_t* __restrict__ A,      // A base + m0*H for this phase
    const uint16_t* __restrict__ curB,   // B source for this phase's prefetches
    const uint16_t* __restrict__ nxtB,   // first-chunk source of next phase (nullptr if none)
    int N0, int ncol, int lane, int tid,
    uint8_t (*ldsb)[BTILE_BYTES],
    v8f& acc0, v8f& acc1)
{
    for (int ch = 0; ch < H / KC; ++ch) {            // 16 chunks
        wait_async0();
        __syncthreads();
        int buf = ch & 1;
        if (ch + 1 < H / KC)
            async_copy_b_tile(curB, N0, (ch + 1) * KC, H - 1, H, ldsb[buf ^ 1], tid);
        else if (nxtB)
            async_copy_b_tile(nxtB, N0, 0, H - 1, H, ldsb[buf ^ 1], tid);
        int k0 = ch * KC;
        #pragma unroll
        for (int ks = 0; ks < KC; ks += 32) {
            v16bf a  = load_frag_a(A, H, k0 + ks, lane);
            v16bf b0 = lds_frag_b(ldsb[buf], ncol, ks, lane);
            v16bf b1 = lds_frag_b(ldsb[buf], ncol + 16, ks, lane);
            acc0 = __builtin_amdgcn_wmma_f32_16x16x32_bf16(false, a, false, b0, (short)0, acc0, false, false);
            acc1 = __builtin_amdgcn_wmma_f32_16x16x32_bf16(false, a, false, b1, (short)0, acc1, false, false);
        }
    }
}

// ---- one-time fp32 -> transposed bf16 weight conversion ----
// in: [batch][K][N] fp32 ; out: [batch][N][K] bf16
__global__ __launch_bounds__(256) void k_transpose_bf16(const float* __restrict__ in,
                                                        uint16_t* __restrict__ out,
                                                        int K, int N) {
    __shared__ float tile[32][33];
    const float* src = in + (size_t)blockIdx.z * K * N;
    uint16_t* dst = out + (size_t)blockIdx.z * (size_t)K * N;
    int n0 = blockIdx.x * 32, k0 = blockIdx.y * 32;
    int tx = threadIdx.x, ty = threadIdx.y;                  // block (32,8)
    #pragma unroll
    for (int r = 0; r < 4; ++r) {
        int k = k0 + ty + r * 8, n = n0 + tx;
        if (k < K && n < N) tile[ty + r * 8][tx] = src[(size_t)k * N + n];
    }
    __syncthreads();
    #pragma unroll
    for (int r = 0; r < 4; ++r) {
        int n = n0 + ty + r * 8, k = k0 + tx;
        if (n < N && k < K) dst[(size_t)n * K + k] = f2bf(tile[tx][ty + r * 8]);
    }
}

__global__ __launch_bounds__(256) void k_zero(float* p, int n) {
    int i = blockIdx.x * 256 + threadIdx.x;
    if (i < n) p[i] = 0.0f;
}

// ---- gates GEMM: g[4][B][H] = xm@Wx + hm@Wh + bias (pre-masked bf16 A-matrices) ----
// WG: 8 waves -> M=64 x N=64 tile; wave = (m-tile 0..3) x (n-pair 0..1), 2 acc/wave.
__global__ __launch_bounds__(256) void k_gates_gemm(
    const uint16_t* __restrict__ xm, const uint16_t* __restrict__ hm,   // [4][B][H] bf16
    const uint16_t* __restrict__ WxT, const uint16_t* __restrict__ WhT, // [4][H][H] bf16 (n-major)
    const float* __restrict__ bias,                                      // [4][H]
    float* __restrict__ gout)                                            // [4][B][H]
{
    __shared__ __align__(16) uint8_t ldsb[2][BTILE_BYTES];
    int tid = threadIdx.x, lane = tid & 31, wave = tid >> 5;
    int gate = blockIdx.y;
    int m0 = (wave & 3) * 16;
    int ncol = (wave >> 2) * 32;
    int N0 = blockIdx.x * 64;

    const uint16_t* Axm = xm + (size_t)gate * BH + (size_t)m0 * H;
    const uint16_t* Ahm = hm + (size_t)gate * BH + (size_t)m0 * H;
    const uint16_t* Bx  = WxT + (size_t)gate * H * H;
    const uint16_t* Bh  = WhT + (size_t)gate * H * H;

    v8f acc0 = {}, acc1 = {};
    async_copy_b_tile(Bx, N0, 0, H - 1, H, ldsb[0], tid);

    gemm_phase(Axm, Bx, Bh,      N0, ncol, lane, tid, ldsb, acc0, acc1);  // 16 chunks (even)
    gemm_phase(Ahm, Bh, nullptr, N0, ncol, lane, tid, ldsb, acc0, acc1);  // parity preserved

    int col = lane & 15, sel = lane >> 4;
    int nA = N0 + ncol + col, nB = nA + 16;
    float biaA = bias[gate * H + nA];
    float biaB = bias[gate * H + nB];
    float* opA = gout + (size_t)gate * BH + (size_t)(m0 + sel * 8) * H + nA;
    float* opB = gout + (size_t)gate * BH + (size_t)(m0 + sel * 8) * H + nB;
    #pragma unroll
    for (int r = 0; r < 8; ++r) {
        opA[(size_t)r * H] = acc0[r] + biaA;
        opB[(size_t)r * H] = acc1[r] + biaB;
    }
}

// ---- decoder GEMM: out[:,t,:] = decA[64,1024] @ dec_W + dec_b ----
__global__ __launch_bounds__(256) void k_dec_gemm(
    const uint16_t* __restrict__ decA, const uint16_t* __restrict__ decWT,
    const float* __restrict__ dec_b, float* __restrict__ out, int t)
{
    __shared__ __align__(16) uint8_t ldsb[2][BTILE_BYTES];
    int tid = threadIdx.x, lane = tid & 31, wave = tid >> 5;
    int m0 = (wave & 3) * 16;
    int ncol = (wave >> 2) * 32;
    int N0 = blockIdx.x * 64;

    const uint16_t* A = decA + (size_t)m0 * H;
    v8f acc0 = {}, acc1 = {};
    async_copy_b_tile(decWT, N0, 0, O - 1, H, ldsb[0], tid);

    const int NCH = H / KC;                          // 16 chunks
    for (int ch = 0; ch < NCH; ++ch) {
        wait_async0();
        __syncthreads();
        int buf = ch & 1;
        if (ch + 1 < NCH)
            async_copy_b_tile(decWT, N0, (ch + 1) * KC, O - 1, H, ldsb[buf ^ 1], tid);
        int k0 = ch * KC;
        #pragma unroll
        for (int ks = 0; ks < KC; ks += 32) {
            v16bf a  = load_frag_a(A, H, k0 + ks, lane);
            v16bf b0 = lds_frag_b(ldsb[buf], ncol, ks, lane);
            v16bf b1 = lds_frag_b(ldsb[buf], ncol + 16, ks, lane);
            acc0 = __builtin_amdgcn_wmma_f32_16x16x32_bf16(false, a, false, b0, (short)0, acc0, false, false);
            acc1 = __builtin_amdgcn_wmma_f32_16x16x32_bf16(false, a, false, b1, (short)0, acc1, false, false);
        }
    }

    int col = lane & 15, sel = lane >> 4;
    int nA = N0 + ncol + col, nB = nA + 16;
    float* rowp = out + (size_t)(m0 + sel * 8) * (T * O) + (size_t)t * O;
    if (nA < O) {
        float bia = dec_b[nA];
        #pragma unroll
        for (int r = 0; r < 8; ++r) rowp[(size_t)r * T * O + nA] = acc0[r] + bia;
    }
    if (nB < O) {
        float bia = dec_b[nB];
        #pragma unroll
        for (int r = 0; r < 8; ++r) rowp[(size_t)r * T * O + nB] = acc1[r] + bia;
    }
}

// ---- prep for t=0, layer 0 (h0 == 0 at this point) ----
__global__ __launch_bounds__(256) void k_prep0(
    const int* __restrict__ ids, const float* __restrict__ emb,
    const float* __restrict__ mask_layer, const float* __restrict__ mask_time,
    const float* __restrict__ hst, uint16_t* __restrict__ xm0, uint16_t* __restrict__ hm0, int t)
{
    int idx = blockIdx.x * 256 + threadIdx.x;     // over B*H
    int b = idx >> 10, hh = idx & (H - 1);
    int id = ids[b * T + t];
    float x = emb[(size_t)id * H + hh];
    float h0 = hst[idx];
    #pragma unroll
    for (int g = 0; g < 4; ++g) {
        xm0[g * BH + idx] = f2bf(x * mask_layer[g * BH + idx]);
        float src = (g < 3) ? h0 : x;             // reference "bug": gate c uses x
        hm0[g * BH + idx] = f2bf(src * mask_time[g * BH + idx]);
    }
}

// ---- pointwise layer0 + prep layer1 A-matrices ----
__global__ __launch_bounds__(256) void k_pw0(
    const float* __restrict__ gbuf, float* __restrict__ hst, float* __restrict__ cst,
    const float* __restrict__ mask_layer, const float* __restrict__ mask_time,
    uint16_t* __restrict__ xm1, uint16_t* __restrict__ hm1)
{
    int idx = blockIdx.x * 256 + threadIdx.x;
    float f  = sigmoidf(gbuf[0 * BH + idx]);
    float i  = sigmoidf(gbuf[1 * BH + idx]);
    float o  = sigmoidf(gbuf[2 * BH + idx]);
    float ct = tanhf(gbuf[3 * BH + idx]);
    float cn = f * cst[idx] + i * ct;
    float hn = o * tanhf(cn);
    cst[idx] = cn;
    hst[idx] = hn;

    float h1_old = hst[BH + idx];
    #pragma unroll
    for (int g = 0; g < 4; ++g) {
        xm1[g * BH + idx] = f2bf(hn * mask_layer[(4 + g) * BH + idx]);
        float src = (g < 3) ? h1_old : hn;
        hm1[g * BH + idx] = f2bf(src * mask_time[(4 + g) * BH + idx]);
    }
}

// ---- pointwise layer1 + decoder-A prep + next-step layer0 prep ----
__global__ __launch_bounds__(256) void k_pw1(
    const float* __restrict__ gbuf, float* __restrict__ hst, float* __restrict__ cst,
    const int* __restrict__ ids, const float* __restrict__ emb,
    const float* __restrict__ mask_layer, const float* __restrict__ mask_time,
    const float* __restrict__ mask_final,
    uint16_t* __restrict__ xm0, uint16_t* __restrict__ hm0,
    uint16_t* __restrict__ decA, int t_next)
{
    int idx = blockIdx.x * 256 + threadIdx.x;
    int b = idx >> 10, hh = idx & (H - 1);
    float f  = sigmoidf(gbuf[0 * BH + idx]);
    float i  = sigmoidf(gbuf[1 * BH + idx]);
    float o  = sigmoidf(gbuf[2 * BH + idx]);
    float ct = tanhf(gbuf[3 * BH + idx]);
    float cn = f * cst[BH + idx] + i * ct;
    float hn = o * tanhf(cn);
    cst[BH + idx] = cn;
    hst[BH + idx] = hn;

    decA[idx] = f2bf(hn * mask_final[idx]);

    if (t_next < T) {
        int id = ids[b * T + t_next];
        float x = emb[(size_t)id * H + hh];
        float h0 = hst[idx];                       // layer0 h already updated this step
        #pragma unroll
        for (int g = 0; g < 4; ++g) {
            xm0[g * BH + idx] = f2bf(x * mask_layer[g * BH + idx]);
            float src = (g < 3) ? h0 : x;
            hm0[g * BH + idx] = f2bf(src * mask_time[g * BH + idx]);
        }
    }
}

extern "C" void kernel_launch(void* const* d_in, const int* in_sizes, int n_in,
                              void* d_out, int out_size, void* d_ws, size_t ws_size,
                              hipStream_t stream) {
    (void)in_sizes; (void)n_in; (void)out_size; (void)ws_size;
    const int*   ids        = (const int*)d_in[0];
    const float* emb        = (const float*)d_in[1];
    const float* Wx         = (const float*)d_in[2];
    const float* bx         = (const float*)d_in[3];
    const float* Wh         = (const float*)d_in[4];
    const float* dec_W      = (const float*)d_in[5];
    const float* dec_b      = (const float*)d_in[6];
    const float* mask_layer = (const float*)d_in[7];
    const float* mask_final = (const float*)d_in[8];
    const float* mask_time  = (const float*)d_in[9];
    float* out = (float*)d_out;

    // workspace carve-up (256B aligned)
    char* base = (char*)d_ws;
    size_t off = 0;
    auto carve = [&](size_t bytes) -> char* {
        char* p = base + off;
        off = (off + bytes + 255) & ~(size_t)255;
        return p;
    };
    uint16_t* WxT   = (uint16_t*)carve((size_t)L * 4 * H * H * 2);   // 16.8 MB
    uint16_t* WhT   = (uint16_t*)carve((size_t)L * 4 * H * H * 2);   // 16.8 MB
    uint16_t* decWT = (uint16_t*)carve((size_t)O * H * 2);           // 20.5 MB
    float*    hst   = (float*)carve((size_t)2 * L * BH * 4);         // h then c
    float*    cst   = hst + (size_t)L * BH;
    uint16_t* xm0   = (uint16_t*)carve((size_t)4 * BH * 2);
    uint16_t* hm0   = (uint16_t*)carve((size_t)4 * BH * 2);
    uint16_t* xm1   = (uint16_t*)carve((size_t)4 * BH * 2);
    uint16_t* hm1   = (uint16_t*)carve((size_t)4 * BH * 2);
    float*    gbuf  = (float*)carve((size_t)4 * BH * 4);
    uint16_t* decA  = (uint16_t*)carve((size_t)BH * 2);

    dim3 tb(32, 8);
    // one-time (per call) weight conversion, transposed to [N][K] bf16
    k_transpose_bf16<<<dim3(H / 32, H / 32, L * 4), tb, 0, stream>>>(Wx, WxT, H, H);
    k_transpose_bf16<<<dim3(H / 32, H / 32, L * 4), tb, 0, stream>>>(Wh, WhT, H, H);
    k_transpose_bf16<<<dim3((O + 31) / 32, H / 32, 1), tb, 0, stream>>>(dec_W, decWT, H, O);

    k_zero<<<(2 * L * BH + 255) / 256, 256, 0, stream>>>(hst, 2 * L * BH);
    k_prep0<<<BH / 256, 256, 0, stream>>>(ids, emb, mask_layer, mask_time, hst, xm0, hm0, 0);

    const size_t gmat = (size_t)4 * H * H;   // per-layer weight block (bf16 elems)
    for (int t = 0; t < T; ++t) {
        k_gates_gemm<<<dim3(H / 64, 4), 256, 0, stream>>>(xm0, hm0, WxT, WhT, bx, gbuf);
        k_pw0<<<BH / 256, 256, 0, stream>>>(gbuf, hst, cst, mask_layer, mask_time, xm1, hm1);
        k_gates_gemm<<<dim3(H / 64, 4), 256, 0, stream>>>(xm1, hm1, WxT + gmat, WhT + gmat,
                                                          bx + 4 * H, gbuf);
        k_pw1<<<BH / 256, 256, 0, stream>>>(gbuf, hst, cst, ids, emb, mask_layer, mask_time,
                                            mask_final, xm0, hm0, decA, t + 1);
        k_dec_gemm<<<dim3((O + 63) / 64, 1), 256, 0, stream>>>(decA, decWT, dec_b, out, t);
    }
}
